// Model_6966436954308
// MI455X (gfx1250) — matrix operand order repaired and checked by reading, compiled, not run on hardware
//
#include <hip/hip_runtime.h>
#include <hip/hip_bf16.h>

// Mini-transformer forward for MI455X (gfx1250, wave32, WMMA + TDM).
// B=32 T=512 VOCAB=128 D=256 H=32 HS=8 ; N = B*T = 16384 tokens.
//
// v3 changes vs v2:
//  - GEMM column-tile loops fully unrolled with DOUBLE-BUFFERED B fragments
//    and __builtin_amdgcn_sched_barrier(0) so next-tile global_load_b128s are
//    issued before the current 8-WMMA chain (round-2 asm showed the scheduler
//    re-serializing to load -> s_wait_loadcnt 0 -> wmma with one reg octet).
//  - prefetch moved two tiles ahead, locality raised.

#define Bb 32
#define Tt 512
#define VOC 128
#define Dd 256
#define Hh 32
#define HS 8
#define Nn (Bb * Tt)        // 16384 tokens

typedef _Float16 half_t;
typedef __attribute__((ext_vector_type(16))) _Float16 v16h;
typedef __attribute__((ext_vector_type(2)))  _Float16 v2h;
typedef __attribute__((ext_vector_type(8)))  float    v8f;
typedef __attribute__((ext_vector_type(4)))  unsigned int u32x4;
typedef __attribute__((ext_vector_type(8)))  int i32x8;
typedef __attribute__((ext_vector_type(4)))  int i32x4;

#ifndef __has_builtin
#define __has_builtin(x) 0
#endif
#if __has_builtin(__builtin_amdgcn_tensor_load_to_lds)
#define HAVE_TDM 1
#else
#define HAVE_TDM 0
#endif

#if __has_builtin(__builtin_amdgcn_sched_barrier)
#define SCHED_FENCE() __builtin_amdgcn_sched_barrier(0)
#else
#define SCHED_FENCE() asm volatile("" ::: "memory")
#endif

static __device__ __forceinline__ v8f wmma16(v16h a, v16h b, v8f c) {
  // 8 args: (neg_a, A, neg_b, B, c_mod, C, reuse_a, reuse_b)
  return __builtin_amdgcn_wmma_f32_16x16x32_f16(false, a, false, b, (short)0, c,
                                                false, false);
}

// ---- fragment loaders (ISA 7.12.2 layouts, wave32) -------------------------
// A-matrix 16x32 f16: lane m%16 holds row m; VGPR j holds K pair
//   k = (j<4?0:16) + 2*(j&3) + 8*(lane/16) (+0/+1). Pairs contiguous -> b128.
static __device__ __forceinline__ v16h load_afrag(const half_t* __restrict__ A,
                                                  int lda, int lane) {
  const int m = lane & 15, hf = lane >> 4;
  v16h a;
#pragma unroll
  for (int j = 0; j < 8; ++j) {
    const int k = ((j & 4) << 2) + 2 * (j & 3) + 8 * hf;
    v2h p = *(const v2h*)(A + (size_t)m * lda + k);
    a[2 * j] = p[0]; a[2 * j + 1] = p[1];
  }
  return a;
}

// B-matrix 32x16 f16 from a COLUMN-MAJOR weight matrix (K contiguous per col):
// lane n%16 holds column n; VGPR j holds K pair k = (j<4?0:16) + 2*(j&3) + 8*(lane/16) (+0/+1).
static __device__ __forceinline__ v16h load_bfrag_cm(const half_t* __restrict__ Bt,
                                                     int ldb, int lane) {
  const int n = lane & 15, kh = lane >> 4;
  v16h b;
#pragma unroll
  for (int j = 0; j < 8; ++j) {
    const int k = ((j & 4) << 2) + 2 * (j & 3) + 8 * kh;
    v2h p = *(const v2h*)(Bt + (size_t)n * ldb + k);
    b[2 * j] = p[0]; b[2 * j + 1] = p[1];
  }
  return b;
}

// ---- TDM staging: copy a 16x256 f16 row-major tile (stride 256) to LDS -----
static __device__ __forceinline__ void stage_tile_16x256(half_t* sA,
                                                         const half_t* gsrc,
                                                         int tid) {
#if HAVE_TDM
  if ((tid >> 5) == 0) {   // one wave issues the TDM op (EXEC ignored by TDM)
    const unsigned long long ga = (unsigned long long)(uintptr_t)gsrc;
    const unsigned int ldsoff = (unsigned int)(uintptr_t)sA; // low 32b = LDS addr
    // D# group0: count=1 | lds_addr | global_addr[56:0] | type=2
    u32x4 g0 = { 1u, ldsoff, (unsigned int)ga,
                 (unsigned int)(((ga >> 32) & 0x1FFFFFFull) | 0x80000000u) };
    // D# group1: data_size=2B; tensor_dim0=256; tensor_dim1=16;
    //            tile_dim0=256; tile_dim1=16; tensor_dim0_stride=256
    i32x8 g1 = { 0x00010000, (int)(256u << 16), (int)(16u << 16),
                 (int)(256u << 16), 16, 256, 0, 0 };
    i32x4 g2 = { 0, 0, 0, 0 };
#if __clang_major__ >= 23
    i32x8 g3z = {};
    __builtin_amdgcn_tensor_load_to_lds(g0, g1, g2, g2, g3z, 0);
#else
    __builtin_amdgcn_tensor_load_to_lds(g0, g1, g2, g2, 0);
#endif
#if __has_builtin(__builtin_amdgcn_s_wait_tensorcnt)
    __builtin_amdgcn_s_wait_tensorcnt(0);
#else
    asm volatile("s_wait_tensorcnt 0x0" ::: "memory");
#endif
  }
  __syncthreads();
#else
  for (int i = tid; i < (16 * 256) / 8; i += 256)
    ((float4*)sA)[i] = ((const float4*)gsrc)[i];
  __syncthreads();
#endif
}

// 16-lane half-group butterfly reductions (wave32: masks 1..8 stay in half)
static __device__ __forceinline__ float redmax16(float v) {
#pragma unroll
  for (int d = 1; d < 16; d <<= 1) v = fmaxf(v, __shfl_xor(v, d, 32));
  return v;
}
static __device__ __forceinline__ float redsum16(float v) {
#pragma unroll
  for (int d = 1; d < 16; d <<= 1) v += __shfl_xor(v, d, 32);
  return v;
}

// ---- kernel 0: repack weights to f16 column-major ---------------------------
__global__ __launch_bounds__(256) void pack_weights(
    const float* __restrict__ Wq, const float* __restrict__ Wk,
    const float* __restrict__ Wv, const float* __restrict__ ffw,
    const float* __restrict__ outw, half_t* __restrict__ wqkv,
    half_t* __restrict__ ffcm, half_t* __restrict__ outcm) {
  int i = blockIdx.x * 256 + threadIdx.x;
  if (i < 3 * Dd * Dd) {                       // QKV: wqkv[col][c]
    int col = i >> 8, c = i & 255;
    int sel = col >> 8, h = (col >> 3) & 31, d = col & 7;
    const float* W = (sel == 0) ? Wq : (sel == 1) ? Wk : Wv;
    wqkv[(size_t)col * Dd + c] = (half_t)W[((size_t)h * Dd + c) * HS + d];
    return;
  }
  i -= 3 * Dd * Dd;
  if (i < Dd * Dd) {                           // ff_w [cin,cout] -> cm
    int cout = i >> 8, cin = i & 255;
    ffcm[(size_t)cout * Dd + cin] = (half_t)ffw[(size_t)cin * Dd + cout];
    return;
  }
  i -= Dd * Dd;
  if (i < VOC * Dd) {                          // out_w [c,v] -> cm
    int v = i >> 8, c = i & 255;
    outcm[(size_t)v * Dd + c] = (half_t)outw[(size_t)c * VOC + v];
  }
}

// ---- kernel 1: embedding -> f16 --------------------------------------------
__global__ __launch_bounds__(256) void embed_kernel(
    const int* __restrict__ idx, const float* __restrict__ tok,
    const float* __restrict__ pos, half_t* __restrict__ xh) {
  int i = blockIdx.x * 256 + threadIdx.x;      // < N*D
  int n = i >> 8, c = i & 255;
  int t = n & (Tt - 1);
  xh[i] = (half_t)(tok[(size_t)idx[n] * Dd + c] + pos[(size_t)t * Dd + c]);
}

// ---- kernel 2: QKV GEMM [N,256]x[256,768], scatter per-head -----------------
// Q,K stored row-major [bh][512][8]; V stored transposed [bh][8][512].
__global__ __launch_bounds__(256) void gemm_qkv(
    const half_t* __restrict__ xh, const half_t* __restrict__ wcm,
    half_t* __restrict__ qh, half_t* __restrict__ kh, half_t* __restrict__ vth) {
  __shared__ __align__(16) half_t sA[16 * Dd];
  const int rt = blockIdx.x;                   // 16-token row tile
  const int w = threadIdx.x >> 5, lane = threadIdx.x & 31;
  const int m16 = lane & 15, hf = lane >> 4;
  stage_tile_16x256(sA, xh + (size_t)rt * 16 * Dd, threadIdx.x);  // TDM
  v16h afr[8];                                 // whole 16x256 A strip in regs
#pragma unroll
  for (int ks = 0; ks < 8; ++ks)
    afr[ks] = load_afrag(sA + ks * 32, Dd, lane);   // from LDS

  v16h bf[2][8];                               // double-buffered B fragments
#pragma unroll
  for (int ks = 0; ks < 8; ++ks)               // prime buffer 0 (ct = w)
    bf[0][ks] = load_bfrag_cm(wcm + (size_t)w * 16 * Dd + ks * 32, Dd, lane);

#pragma unroll
  for (int it = 0; it < 6; ++it) {             // 6 col tiles per wave
    const int ct = w + it * 8;
    if (it + 2 < 6)
      __builtin_prefetch(wcm + (size_t)(ct + 16) * 16 * Dd, 0, 1);
    if (it + 1 < 6) {
#pragma unroll
      for (int ks = 0; ks < 8; ++ks)           // next tile: 16 b128 in flight
        bf[(it + 1) & 1][ks] =
            load_bfrag_cm(wcm + (size_t)(ct + 8) * 16 * Dd + ks * 32, Dd, lane);
    }
    SCHED_FENCE();                             // keep loads above the WMMAs
    v8f c = {};
#pragma unroll
    for (int ks = 0; ks < 8; ++ks) c = wmma16(afr[ks], bf[it & 1][ks], c);
    const int gcol = ct * 16 + m16;
    const int sel = gcol >> 8, h = (gcol >> 3) & 31, d = gcol & 7;
#pragma unroll
    for (int r = 0; r < 8; ++r) {
      const int token = rt * 16 + r + 8 * hf;  // C layout: M = r + 8*(lane/16)
      const int b_ = token >> 9, t = token & (Tt - 1);
      const int bh = b_ * Hh + h;
      const half_t val = (half_t)c[r];
      if (sel == 0)      qh[((size_t)bh * Tt + t) * HS + d] = val;
      else if (sel == 1) kh[((size_t)bh * Tt + t) * HS + d] = val;
      else               vth[((size_t)bh * HS + d) * Tt + t] = val;
    }
  }
}

// ---- kernel 3: flash attention per (b,h); WMMA QK^T and P.V -----------------
__global__ __launch_bounds__(256) void attn_kernel(
    const half_t* __restrict__ qh, const half_t* __restrict__ kh,
    const half_t* __restrict__ vth, half_t* __restrict__ yh) {
  __shared__ __align__(16) half_t sP[8][16][32]; // wave-private P staging
  const int bh = blockIdx.x;
  const int w = threadIdx.x >> 5, lane = threadIdx.x & 31;
  const int m16 = lane & 15, hf = lane >> 4;
  const half_t* Q  = qh  + (size_t)bh * Tt * HS;
  const half_t* K  = kh  + (size_t)bh * Tt * HS;
  const half_t* Vt = vth + (size_t)bh * HS * Tt;
  const int hhead = bh & (Hh - 1), bb = bh >> 5;

  for (int qt = w; qt < Tt / 16; qt += 8) {    // 4 query tiles per wave
    // Q as padded A-frag 16x32 (only k<8 valid -> lanes 0-15, VGPR 0-3)
    v16h qa = {};
    if (hf == 0) {
#pragma unroll
      for (int j = 0; j < 4; ++j) {
        v2h p = *(const v2h*)(Q + ((size_t)qt * 16 + m16) * HS + 2 * j);
        qa[2 * j] = p[0]; qa[2 * j + 1] = p[1];
      }
    }
    v8f oacc = {};
    float mrow[8], lrow[8];
#pragma unroll
    for (int r = 0; r < 8; ++r) { mrow[r] = -1e30f; lrow[r] = 0.f; }

    const int ncc = (qt >> 1) + 1;             // 32-key chunks (causal)
    for (int cc = 0; cc < ncc; ++cc) {
      const int kt0 = cc * 2, kt1 = cc * 2 + 1;
      // K tiles as padded B-frags: B[k=hs][n=key], valid hs<8
      v16h kb0 = {}, kb1 = {};
      if (hf == 0) {
#pragma unroll
        for (int j = 0; j < 4; ++j) {
          v2h p0 = *(const v2h*)(K + ((size_t)kt0 * 16 + m16) * HS + 2 * j);
          v2h p1 = *(const v2h*)(K + ((size_t)kt1 * 16 + m16) * HS + 2 * j);
          kb0[2 * j] = p0[0]; kb0[2 * j + 1] = p0[1];
          kb1[2 * j] = p1[0]; kb1[2 * j + 1] = p1[1];
        }
      }
      v8f z = {};
      v8f s0 = wmma16(qa, kb0, z);
      v8f s1 = wmma16(qa, kb1, z);

      // online softmax; NOTE reference scales by D**-0.5 = 1/16
#pragma unroll
      for (int r = 0; r < 8; ++r) {
        const int qrow = qt * 16 + r + 8 * hf;
        const float x0 = (kt0 * 16 + m16 <= qrow) ? s0[r] * 0.0625f : -1e30f;
        const float x1 = (kt1 * 16 + m16 <= qrow) ? s1[r] * 0.0625f : -1e30f;
        const float cmax = redmax16(fmaxf(x0, x1));
        const float newm = fmaxf(mrow[r], cmax);
        const float p0 = __expf(x0 - newm);
        const float p1 = __expf(x1 - newm);
        const float psum = redsum16(p0 + p1);
        const float alpha = __expf(mrow[r] - newm);
        lrow[r] = lrow[r] * alpha + psum;
        mrow[r] = newm;
        oacc[r] *= alpha;
        sP[w][r + 8 * hf][m16]      = (half_t)p0;   // row-major [16][32]
        sP[w][r + 8 * hf][16 + m16] = (half_t)p1;
      }
      asm volatile("s_wait_dscnt 0" ::: "memory");  // C-layout -> A-layout swap

      v16h pa;                                  // P as A-frag from LDS
#pragma unroll
      for (int j = 0; j < 8; ++j) {
        const int k = ((j & 4) << 2) + 2 * (j & 3) + 8 * hf;
        v2h p = *(const v2h*)&sP[w][m16][k];
        pa[2 * j] = p[0]; pa[2 * j + 1] = p[1];
      }
      // V chunk as B-frag: B[k=key 0..31][n=hs], valid n<8; Vt is [8][512]
      v16h vb = {};
      if (m16 < 8) {
#pragma unroll
        for (int j = 0; j < 8; ++j) {
          const int kloc = ((j & 4) << 2) + 2 * (j & 3) + 8 * hf;
          v2h p = *(const v2h*)(Vt + (size_t)m16 * Tt + cc * 32 + kloc);
          vb[2 * j] = p[0]; vb[2 * j + 1] = p[1];
        }
      }
      oacc = wmma16(pa, vb, oacc);
      asm volatile("" ::: "memory");            // keep next stores after reads
    }
    // normalize + store concatenated heads: yh[token][h*8 + d]
    if (m16 < 8) {
#pragma unroll
      for (int r = 0; r < 8; ++r) {
        const int qrow = qt * 16 + r + 8 * hf;
        const int token = bb * Tt + qrow;
        yh[(size_t)token * Dd + hhead * HS + m16] = (half_t)(oacc[r] / lrow[r]);
      }
    }
  }
}

// ---- kernel 4: FF GEMM [N,256]x[256,256] + bias + ReLU ----------------------
__global__ __launch_bounds__(256) void gemm_ff(
    const half_t* __restrict__ yh, const half_t* __restrict__ wcm,
    const float* __restrict__ bias, half_t* __restrict__ zh) {
  __shared__ __align__(16) half_t sA[16 * Dd];
  const int rt = blockIdx.x;
  const int w = threadIdx.x >> 5, lane = threadIdx.x & 31;
  const int m16 = lane & 15, hf = lane >> 4;
  stage_tile_16x256(sA, yh + (size_t)rt * 16 * Dd, threadIdx.x);  // TDM
  v16h afr[8];
#pragma unroll
  for (int ks = 0; ks < 8; ++ks)
    afr[ks] = load_afrag(sA + ks * 32, Dd, lane);

  v16h bf[2][8];
#pragma unroll
  for (int ks = 0; ks < 8; ++ks)               // prime buffer 0 (ct = w)
    bf[0][ks] = load_bfrag_cm(wcm + (size_t)w * 16 * Dd + ks * 32, Dd, lane);
#pragma unroll
  for (int it = 0; it < 2; ++it) {
    const int ct = w + it * 8;
    if (it + 1 < 2) {
#pragma unroll
      for (int ks = 0; ks < 8; ++ks)
        bf[1][ks] =
            load_bfrag_cm(wcm + (size_t)(ct + 8) * 16 * Dd + ks * 32, Dd, lane);
    }
    SCHED_FENCE();
    v8f c = {};
#pragma unroll
    for (int ks = 0; ks < 8; ++ks) c = wmma16(afr[ks], bf[it & 1][ks], c);
    const int col = ct * 16 + m16;
    const float bv = bias[col];
#pragma unroll
    for (int r = 0; r < 8; ++r) {
      const int row = rt * 16 + r + 8 * hf;
      zh[(size_t)row * Dd + col] = (half_t)fmaxf(c[r] + bv, 0.f);
    }
  }
}

// ---- kernel 5: output GEMM [N,256]x[256,128] + bias -> f32 logits -----------
__global__ __launch_bounds__(256) void gemm_out(
    const half_t* __restrict__ zh, const half_t* __restrict__ wcm,
    const float* __restrict__ bias, float* __restrict__ logits) {
  __shared__ __align__(16) half_t sA[16 * Dd];
  const int rt = blockIdx.x;
  const int w = threadIdx.x >> 5, lane = threadIdx.x & 31;
  const int m16 = lane & 15, hf = lane >> 4;
  stage_tile_16x256(sA, zh + (size_t)rt * 16 * Dd, threadIdx.x);  // TDM
  const int ct = w;                             // 8 col tiles, 1 per wave
  v16h bf[8];
#pragma unroll
  for (int ks = 0; ks < 8; ++ks)
    bf[ks] = load_bfrag_cm(wcm + (size_t)ct * 16 * Dd + ks * 32, Dd, lane);
  v16h afr[8];
#pragma unroll
  for (int ks = 0; ks < 8; ++ks)
    afr[ks] = load_afrag(sA + ks * 32, Dd, lane);
  SCHED_FENCE();
  v8f c = {};
#pragma unroll
  for (int ks = 0; ks < 8; ++ks) c = wmma16(afr[ks], bf[ks], c);
  const int col = ct * 16 + m16;
  const float bv = bias[col];
#pragma unroll
  for (int r = 0; r < 8; ++r) {
    const int row = rt * 16 + r + 8 * hf;
    logits[(size_t)row * VOC + col] = c[r] + bv;
  }
}

// ---- kernels 6/7: deterministic NLL loss (no float atomics) -----------------
__global__ __launch_bounds__(256) void row_loss(
    const float* __restrict__ logits, const int* __restrict__ tgt,
    float* __restrict__ partials) {
  const int w = threadIdx.x >> 5, lane = threadIdx.x & 31;
  const int row = blockIdx.x * 8 + w;
  const float* lr = logits + (size_t)row * VOC;
  float v0 = lr[lane], v1 = lr[lane + 32], v2 = lr[lane + 64], v3 = lr[lane + 96];
  float mx = fmaxf(fmaxf(v0, v1), fmaxf(v2, v3));
#pragma unroll
  for (int d = 1; d < 32; d <<= 1) mx = fmaxf(mx, __shfl_xor(mx, d, 32));
  float s = __expf(v0 - mx) + __expf(v1 - mx) + __expf(v2 - mx) + __expf(v3 - mx);
#pragma unroll
  for (int d = 1; d < 32; d <<= 1) s += __shfl_xor(s, d, 32);
  const float lse = mx + __logf(s);
  const float li = lse - lr[tgt[row]];
  __shared__ float acc[8];
  if (lane == 0) acc[w] = li;
  __syncthreads();
  if (threadIdx.x == 0) {
    float t = 0.f;
#pragma unroll
    for (int i = 0; i < 8; ++i) t += acc[i];
    partials[blockIdx.x] = t;
  }
}

__global__ __launch_bounds__(256) void final_loss(
    const float* __restrict__ partials, float* __restrict__ out_loss) {
  float t = 0.f;
  for (int i = threadIdx.x; i < Nn / 8; i += 256) t += partials[i];
  __shared__ float acc[256];
  acc[threadIdx.x] = t;
  __syncthreads();
  for (int s = 128; s > 0; s >>= 1) {
    if (threadIdx.x < s) acc[threadIdx.x] += acc[threadIdx.x + s];
    __syncthreads();
  }
  if (threadIdx.x == 0) *out_loss = acc[0] / (float)Nn;
}

// ---- host launcher ----------------------------------------------------------
extern "C" void kernel_launch(void* const* d_in, const int* in_sizes, int n_in,
                              void* d_out, int out_size, void* d_ws, size_t ws_size,
                              hipStream_t stream) {
  const int*   idx  = (const int*)  d_in[0];
  const int*   tgt  = (const int*)  d_in[1];
  const float* tok  = (const float*)d_in[2];
  const float* pos  = (const float*)d_in[3];
  const float* Wq   = (const float*)d_in[4];
  const float* Wk   = (const float*)d_in[5];
  const float* Wv   = (const float*)d_in[6];
  const float* ffw  = (const float*)d_in[7];
  const float* ffb  = (const float*)d_in[8];
  const float* outw = (const float*)d_in[9];
  const float* outb = (const float*)d_in[10];
  float* out = (float*)d_out;

  char* p = (char*)d_ws;
  auto alloc = [&](size_t bytes) -> char* {
    char* r = p; p += (bytes + 255) & ~(size_t)255; return r;
  };
  half_t* xh    = (half_t*)alloc((size_t)Nn * Dd * 2);           // 8 MB
  half_t* wqkv  = (half_t*)alloc((size_t)3 * Dd * Dd * 2);       // 384 KB
  half_t* ffcm  = (half_t*)alloc((size_t)Dd * Dd * 2);           // 128 KB
  half_t* outcm = (half_t*)alloc((size_t)VOC * Dd * 2);          // 64 KB
  half_t* qh    = (half_t*)alloc((size_t)Bb * Hh * Tt * HS * 2); // 8 MB
  half_t* kh    = (half_t*)alloc((size_t)Bb * Hh * Tt * HS * 2); // 8 MB
  half_t* vth   = (half_t*)alloc((size_t)Bb * Hh * HS * Tt * 2); // 8 MB
  half_t* yh    = (half_t*)alloc((size_t)Nn * Dd * 2);           // 8 MB
  half_t* zh    = (half_t*)alloc((size_t)Nn * Dd * 2);           // 8 MB
  float*  part  = (float*)alloc((size_t)(Nn / 8) * 4);           // 8 KB
  (void)ws_size; (void)in_sizes; (void)n_in; (void)out_size;

  pack_weights<<<(3 * Dd * Dd + Dd * Dd + VOC * Dd + 255) / 256, 256, 0, stream>>>(
      Wq, Wk, Wv, ffw, outw, wqkv, ffcm, outcm);
  embed_kernel<<<Nn * Dd / 256, 256, 0, stream>>>(idx, tok, pos, xh);
  gemm_qkv<<<Nn / 16, 256, 0, stream>>>(xh, wqkv, qh, kh, vth);
  attn_kernel<<<Bb * Hh, 256, 0, stream>>>(qh, kh, vth, yh);
  gemm_ff<<<Nn / 16, 256, 0, stream>>>(yh, ffcm, ffb, zh);
  gemm_out<<<Nn / 16, 256, 0, stream>>>(zh, outcm, outb, out);
  row_loss<<<Nn / 8, 256, 0, stream>>>(out, tgt, part);
  final_loss<<<1, 256, 0, stream>>>(part, out + (size_t)Nn * VOC);
}
